// ABL_EDGE_62560493634183
// MI455X (gfx1250) — compile-verified, matrix operand order checked by reading, not run on hardware
//
#include <hip/hip_runtime.h>
#include <hip/hip_bf16.h>

typedef __attribute__((ext_vector_type(16))) _Float16 v16h;
typedef __attribute__((ext_vector_type(8)))  _Float16 v8h;
typedef __attribute__((ext_vector_type(8)))  float    v8f;
typedef __attribute__((ext_vector_type(4)))  unsigned int v4u;
typedef __attribute__((ext_vector_type(8)))  int v8i;
typedef __attribute__((ext_vector_type(4)))  int v4i;

static constexpr int   NNODE  = 20;     // N
static constexpr int   NC2    = 190;    // N*(N-1)/2
static constexpr float TEMP_  = 0.1f;
static constexpr int   ITERS_ = 20;

#if defined(__has_builtin)
#if __has_builtin(__builtin_amdgcn_tensor_load_to_lds) && __has_builtin(__builtin_amdgcn_s_wait_tensorcnt)
#define USE_TDM 1
#endif
#endif
#ifndef USE_TDM
#define USE_TDM 0
#endif

// ---------------- WMMA helpers (wave32, 16x16x32 f16 -> f32) ----------------

__device__ __forceinline__ v8f wmma32(v16h a, v16h b, v8f c) {
    return __builtin_amdgcn_wmma_f32_16x16x32_f16(
        false, a, false, b, (short)0, c, false, false);
}

// A fragment: 16x32 f16 tile, row-major in LDS, leading dim lda (halves, mult of 8).
// Two contiguous 16-byte runs per lane -> 2x ds_load_b128.
__device__ __forceinline__ v16h frag_a(const _Float16* A, int lda, int k0) {
    int lane = threadIdx.x & 31;
    int m = lane & 15, hi = lane >> 4;
    const _Float16* p = A + m * lda + k0 + hi * 8;
    v8h lo = *(const v8h*)(p);
    v8h hh = *(const v8h*)(p + 16);
    v16h r = __builtin_shufflevector(lo, hh, 0,1,2,3,4,5,6,7,8,9,10,11,12,13,14,15);
    return r;
}

// Swizzled B buffer: fragment-linear layout, 512 halves per (k-chunk, n-subtile):
//   sBs[ ((c*NT + t)*32 + lane)*16 + e ]
// so each lane reads its whole fragment as 32 contiguous bytes.
__device__ __forceinline__ v16h frag_b_sw(const _Float16* sBs, int NT, int c, int t) {
    int lane = threadIdx.x & 31;
    const _Float16* p = sBs + (((c * NT + t) * 32) + lane) * 16;
    v8h lo = *(const v8h*)(p);
    v8h hh = *(const v8h*)(p + 8);
    v16h r = __builtin_shufflevector(lo, hh, 0,1,2,3,4,5,6,7,8,9,10,11,12,13,14,15);
    return r;
}

// Stage W[Ksrc x Nsrc] (row-major f32, global or LDS) into swizzled f16 layout,
// zero-padded to KC*32 rows x NT*16 cols.
__device__ __forceinline__ void stage_b_sw(_Float16* sBs, const float* W,
                                           int Ksrc, int Nsrc, int KC, int NT,
                                           int tid, int nthr) {
    int total = KC * NT * 512;
    for (int idx = tid; idx < total; idx += nthr) {
        int e    = idx & 15;
        int lane = (idx >> 4) & 31;
        int rest = idx >> 9;
        int t = rest % NT, c = rest / NT;
        int n = lane & 15, hi = lane >> 4;
        int j = e & 7;
        int k   = c * 32 + ((e < 8) ? 0 : 16) + hi * 8 + j;
        int col = t * 16 + n;
        float v = (k < Ksrc && col < Nsrc) ? W[k * Nsrc + col] : 0.f;
        sBs[idx] = (_Float16)v;
    }
}

// ---------------- Tensor Data Mover: contiguous f32 block -> LDS ----------------
// D# per CDNA5 ISA: 2-D tensor, dim0=nelem,
// dim1=1, tile=whole tensor, data_size=4B, type=2 ("image").
__device__ __forceinline__ void tdm_load_f32(float* lds_dst, const float* gsrc, int nelem) {
#if USE_TDM
    unsigned lds_off = (unsigned)(size_t)(void*)lds_dst;        // LDS byte offset
    unsigned long long ga = (unsigned long long)gsrc;
    unsigned dim0 = (unsigned)nelem;
    v4u g0;
    g0[0] = 1u;                                                 // count=1, user mode
    g0[1] = lds_off;                                            // lds_addr
    g0[2] = (unsigned)(ga & 0xffffffffu);                       // global_addr[31:0]
    g0[3] = (unsigned)((ga >> 32) & 0x01ffffffu) | (2u << 30);  // addr[56:32], type=2
    v8i g1;
    g1[0] = (int)(2u << 16);                        // workgroup_mask=0, data_size=4B
    g1[1] = (int)((dim0 & 0xffffu) << 16);          // tensor_dim0[15:0]  @bits63:48
    g1[2] = (int)((dim0 >> 16) & 0xffffu)           // tensor_dim0[31:16] @bits79:64
          | (int)(1u << 16);                        // tensor_dim1=1      @bits95:80
    g1[3] = (int)((dim0 & 0xffffu) << 16);          // tile_dim0          @bits127:112
    g1[4] = 1;                                      // tile_dim1=1, tile_dim2=0
    g1[5] = (int)dim0;                              // tensor_dim0_stride[31:0]
    g1[6] = 0;
    g1[7] = 0;
    v4i gz4 = {0, 0, 0, 0};
    v8i gz8 = {0, 0, 0, 0, 0, 0, 0, 0};
    __builtin_amdgcn_tensor_load_to_lds(g0, g1, gz4, gz4, gz8, 0);
    __builtin_amdgcn_s_wait_tensorcnt((short)0);
#else
    (void)lds_dst; (void)gsrc; (void)nelem;
#endif
}

__device__ __forceinline__ void triu_ij(int p, int& i, int& j) {
    int i0 = 0, rem = p, cnt = NNODE - 1;
    while (rem >= cnt) { rem -= cnt; --cnt; ++i0; }
    i = i0; j = i0 + 1 + rem;
}

// ---------------- kernels ----------------

__global__ void k_zero(float* __restrict__ p, int n) {
    int i = blockIdx.x * blockDim.x + threadIdx.x;
    if (i < n) p[i] = 0.f;
}

// Y[rows x 64] = X[rows x K] @ W[K x 64] + bias   (K <= 32). 128 thr = 4 waves.
__global__ void k_embed(const float* __restrict__ X, const float* __restrict__ W,
                        const float* __restrict__ bias, int K, float* __restrict__ Y) {
    __shared__ alignas(32) _Float16 sA[16 * 32];
    __shared__ alignas(32) _Float16 sB[1 * 4 * 512];
    int tile = blockIdx.x, tid = threadIdx.x;
    for (int idx = tid; idx < 16 * 32; idx += 128) {
        int m = idx >> 5, k = idx & 31;
        sA[idx] = (k < K) ? (_Float16)X[(tile * 16 + m) * K + k] : (_Float16)0.f;
    }
    stage_b_sw(sB, W, K, 64, 1, 4, tid, 128);
    __syncthreads();
    int wid = tid >> 5, lane = tid & 31;
    int n0 = wid * 16, n = lane & 15, hi = lane >> 4;
    float bv = bias[n0 + n];
    v8f acc; for (int r = 0; r < 8; ++r) acc[r] = bv;
    acc = wmma32(frag_a(sA, 32, 0), frag_b_sw(sB, 4, 0, wid), acc);
    for (int r = 0; r < 8; ++r)
        Y[(tile * 16 + r + 8 * hi) * 64 + n0 + n] = acc[r];
}

// msg = relu([h_from|h_to|e] @ Wm1 + bm1) @ Wm2 + bm2 ; atomicAdd into agg[to].
__global__ void k_message(const float* __restrict__ h, const float* __restrict__ e,
                          const int* __restrict__ from_idx, const int* __restrict__ to_idx,
                          const float* __restrict__ Wm1, const float* __restrict__ bm1,
                          const float* __restrict__ Wm2, const float* __restrict__ bm2,
                          float* __restrict__ agg) {
    __shared__ alignas(32) _Float16 sX[16 * 192];
    __shared__ alignas(32) _Float16 sW1[6 * 4 * 512];
    __shared__ alignas(32) _Float16 sY[16 * 64];
    __shared__ alignas(32) _Float16 sW2[2 * 4 * 512];
#if USE_TDM
    __shared__ alignas(16) float sWf[64 * 64];
#endif
    int tile = blockIdx.x, tid = threadIdx.x;
    __builtin_prefetch(Wm1, 0, 1);
    for (int idx = tid; idx < 16 * 192; idx += 128) {
        int m = idx / 192, k = idx % 192;
        int ei = tile * 16 + m;
        float v;
        if (k < 64)       v = h[from_idx[ei] * 64 + k];
        else if (k < 128) v = h[to_idx[ei] * 64 + (k - 64)];
        else              v = e[ei * 64 + (k - 128)];
        sX[idx] = (_Float16)v;
    }
    stage_b_sw(sW1, Wm1, 192, 64, 6, 4, tid, 128);
#if USE_TDM
    if (tid < 32) tdm_load_f32(sWf, Wm2, 64 * 64);
    __syncthreads();
    stage_b_sw(sW2, sWf, 64, 64, 2, 4, tid, 128);
#else
    stage_b_sw(sW2, Wm2, 64, 64, 2, 4, tid, 128);
#endif
    __syncthreads();
    int wid = tid >> 5, lane = tid & 31;
    int n0 = wid * 16, n = lane & 15, hi = lane >> 4;
    {
        float bv = bm1[n0 + n];
        v8f acc; for (int r = 0; r < 8; ++r) acc[r] = bv;
#pragma unroll
        for (int c = 0; c < 6; ++c)
            acc = wmma32(frag_a(sX, 192, c * 32), frag_b_sw(sW1, 4, c, wid), acc);
        for (int r = 0; r < 8; ++r)
            sY[(r + 8 * hi) * 64 + n0 + n] = (_Float16)fmaxf(acc[r], 0.f);
    }
    __syncthreads();
    {
        float bv = bm2[n0 + n];
        v8f acc; for (int r = 0; r < 8; ++r) acc[r] = bv;
#pragma unroll
        for (int c = 0; c < 2; ++c)
            acc = wmma32(frag_a(sY, 64, c * 32), frag_b_sw(sW2, 4, c, wid), acc);
        for (int r = 0; r < 8; ++r) {
            int ei = tile * 16 + r + 8 * hi;
            atomicAdd(&agg[to_idx[ei] * 64 + n0 + n], acc[r]);
        }
    }
}

// h_new = relu([h|agg] @ Wu1 + bu1) @ Wu2 + bu2
__global__ void k_update(const float* __restrict__ h, const float* __restrict__ agg,
                         const float* __restrict__ Wu1, const float* __restrict__ bu1,
                         const float* __restrict__ Wu2, const float* __restrict__ bu2,
                         float* __restrict__ hn) {
    __shared__ alignas(32) _Float16 sX[16 * 128];
    __shared__ alignas(32) _Float16 sW1[4 * 4 * 512];
    __shared__ alignas(32) _Float16 sY[16 * 64];
    __shared__ alignas(32) _Float16 sW2[2 * 4 * 512];
#if USE_TDM
    __shared__ alignas(16) float sWf[64 * 64];
#endif
    int tile = blockIdx.x, tid = threadIdx.x;
    __builtin_prefetch(Wu1, 0, 1);
    for (int idx = tid; idx < 16 * 128; idx += 128) {
        int m = idx >> 7, k = idx & 127;
        int row = tile * 16 + m;
        float v = (k < 64) ? h[row * 64 + k] : agg[row * 64 + (k - 64)];
        sX[idx] = (_Float16)v;
    }
    stage_b_sw(sW1, Wu1, 128, 64, 4, 4, tid, 128);
#if USE_TDM
    if (tid < 32) tdm_load_f32(sWf, Wu2, 64 * 64);
    __syncthreads();
    stage_b_sw(sW2, sWf, 64, 64, 2, 4, tid, 128);
#else
    stage_b_sw(sW2, Wu2, 64, 64, 2, 4, tid, 128);
#endif
    __syncthreads();
    int wid = tid >> 5, lane = tid & 31;
    int n0 = wid * 16, n = lane & 15, hi = lane >> 4;
    {
        float bv = bu1[n0 + n];
        v8f acc; for (int r = 0; r < 8; ++r) acc[r] = bv;
#pragma unroll
        for (int c = 0; c < 4; ++c)
            acc = wmma32(frag_a(sX, 128, c * 32), frag_b_sw(sW1, 4, c, wid), acc);
        for (int r = 0; r < 8; ++r)
            sY[(r + 8 * hi) * 64 + n0 + n] = (_Float16)fmaxf(acc[r], 0.f);
    }
    __syncthreads();
    {
        float bv = bu2[n0 + n];
        v8f acc; for (int r = 0; r < 8; ++r) acc[r] = bv;
#pragma unroll
        for (int c = 0; c < 2; ++c)
            acc = wmma32(frag_a(sY, 64, c * 32), frag_b_sw(sW2, 4, c, wid), acc);
        for (int r = 0; r < 8; ++r)
            hn[(tile * 16 + r + 8 * hi) * 64 + n0 + n] = acc[r];
    }
}

// T = relu(H @ Ws1 + bs1) @ Ws2 + bs2, scattered into Tq (even graphs) / Tc (odd).
__global__ void k_scores(const float* __restrict__ H,
                         const float* __restrict__ Ws1, const float* __restrict__ bs1,
                         const float* __restrict__ Ws2, const float* __restrict__ bs2,
                         float* __restrict__ Tq, float* __restrict__ Tc) {
    __shared__ alignas(32) _Float16 sX[16 * 64];
    __shared__ alignas(32) _Float16 sW1[2 * 2 * 512];
    __shared__ alignas(32) _Float16 sT[16 * 32];
    __shared__ alignas(32) _Float16 sW2[1 * 2 * 512];
    int tile = blockIdx.x, tid = threadIdx.x;   // 64 threads = 2 waves
    for (int idx = tid; idx < 16 * 64; idx += 64) sX[idx] = (_Float16)H[tile * 16 * 64 + idx];
    stage_b_sw(sW1, Ws1, 64, NNODE, 2, 2, tid, 64);
    stage_b_sw(sW2, Ws2, NNODE, NNODE, 1, 2, tid, 64);
    __syncthreads();
    int wid = tid >> 5, lane = tid & 31;
    int n0 = wid * 16, n = lane & 15, hi = lane >> 4;
    int nc = n0 + n;
    {
        float bv = (nc < NNODE) ? bs1[nc] : 0.f;
        v8f acc; for (int r = 0; r < 8; ++r) acc[r] = bv;
        acc = wmma32(frag_a(sX, 64, 0),  frag_b_sw(sW1, 2, 0, wid), acc);
        acc = wmma32(frag_a(sX, 64, 32), frag_b_sw(sW1, 2, 1, wid), acc);
        for (int r = 0; r < 8; ++r)
            sT[(r + 8 * hi) * 32 + nc] = (_Float16)fmaxf(acc[r], 0.f);
    }
    __syncthreads();
    {
        float bv = (nc < NNODE) ? bs2[nc] : 0.f;
        v8f acc; for (int r = 0; r < 8; ++r) acc[r] = bv;
        acc = wmma32(frag_a(sT, 32, 0), frag_b_sw(sW2, 2, 0, wid), acc);
        if (nc < NNODE) {
            for (int r = 0; r < 8; ++r) {
                int gr = tile * 16 + r + 8 * hi;           // global node row < 1280
                int g = gr / NNODE, i = gr % NNODE;
                float* dst = (g & 1) ? Tc : Tq;
                dst[(g >> 1) * (NNODE * NNODE) + i * NNODE + nc] = acc[r];
            }
        }
    }
}

// cost(i,j) = sum_k |Tq[i,k]-Tc[j,k]|; P = sinkhorn(-cost)
__global__ void k_cost_sinkhorn(const float* __restrict__ Tq, const float* __restrict__ Tc,
                                float* __restrict__ P) {
    __shared__ float sq[400], sc[400], la[400];
    int b = blockIdx.x, tid = threadIdx.x;   // 128
    for (int idx = tid; idx < 400; idx += 128) { sq[idx] = Tq[b * 400 + idx]; sc[idx] = Tc[b * 400 + idx]; }
    __syncthreads();
    for (int idx = tid; idx < 400; idx += 128) {
        int i = idx / NNODE, j = idx % NNODE;
        float c = 0.f;
        for (int k = 0; k < NNODE; ++k) c += fabsf(sq[i * NNODE + k] - sc[j * NNODE + k]);
        la[idx] = -c / TEMP_;
    }
    __syncthreads();
    for (int it = 0; it < ITERS_; ++it) {
        if (tid < NNODE) {
            float m = -INFINITY;
            for (int j = 0; j < NNODE; ++j) m = fmaxf(m, la[tid * NNODE + j]);
            float s = 0.f;
            for (int j = 0; j < NNODE; ++j) s += __expf(la[tid * NNODE + j] - m);
            float lse = m + __logf(s);
            for (int j = 0; j < NNODE; ++j) la[tid * NNODE + j] -= lse;
        }
        __syncthreads();
        if (tid < NNODE) {
            float m = -INFINITY;
            for (int i = 0; i < NNODE; ++i) m = fmaxf(m, la[i * NNODE + tid]);
            float s = 0.f;
            for (int i = 0; i < NNODE; ++i) s += __expf(la[i * NNODE + tid] - m);
            float lse = m + __logf(s);
            for (int i = 0; i < NNODE; ++i) la[i * NNODE + tid] -= lse;
        }
        __syncthreads();
    }
    for (int idx = tid; idx < 400; idx += 128) P[b * 400 + idx] = __expf(la[idx]);
}

// qe/ce: fwd+bwd 2-layer MLP on [s|d|adj] per node pair, WMMA, K padded 129->160.
__global__ void k_pair_emb(const float* __restrict__ H,
                           const float* __restrict__ qadj, const float* __restrict__ cadj,
                           const float* __restrict__ Wl1, const float* __restrict__ bl1,
                           const float* __restrict__ Wl2, const float* __restrict__ bl2,
                           float* __restrict__ qe, float* __restrict__ ce) {
    __shared__ alignas(32) _Float16 sX[16 * 160];
    __shared__ alignas(32) _Float16 sW1[5 * 4 * 512];
    __shared__ alignas(32) _Float16 sY[16 * 64];
    __shared__ alignas(32) _Float16 sW2[2 * 4 * 512];
#if USE_TDM
    __shared__ alignas(16) float sWf[64 * 64];
#endif
    int pblk = blockIdx.x;              // 0..23 (8 pairs each)
    int b    = blockIdx.y;              // 0..31
    int which = blockIdx.z;             // 0 = query, 1 = corpus
    int tid = threadIdx.x;              // 128
    int p0 = pblk * 8;
    int g  = b * 2 + which;
    const float* adj = which ? cadj : qadj;
    float* outp = which ? ce : qe;
    __builtin_prefetch(Wl1, 0, 1);
    for (int idx = tid; idx < 16 * 160; idx += 128) {
        int m = idx / 160, k = idx % 160;
        int pp = m >> 1, dir = m & 1;
        int p = p0 + pp;
        float v = 0.f;
        if (p < NC2) {
            int i, j; triu_ij(p, i, j);
            int an  = dir ? j : i;
            int bn2 = dir ? i : j;
            if (k < 64)        v = H[(g * NNODE + an) * 64 + k];
            else if (k < 128)  v = H[(g * NNODE + bn2) * 64 + (k - 64)];
            else if (k == 128) v = adj[b * NNODE * NNODE + i * NNODE + j];
        }
        sX[idx] = (_Float16)v;
    }
    stage_b_sw(sW1, Wl1, 129, 64, 5, 4, tid, 128);
#if USE_TDM
    if (tid < 32) tdm_load_f32(sWf, Wl2, 64 * 64);
    __syncthreads();
    stage_b_sw(sW2, sWf, 64, 64, 2, 4, tid, 128);
#else
    stage_b_sw(sW2, Wl2, 64, 64, 2, 4, tid, 128);
#endif
    __syncthreads();
    int wid = tid >> 5, lane = tid & 31;
    int n0 = wid * 16, n = lane & 15, hi = lane >> 4;
    {
        float bv = bl1[n0 + n];
        v8f acc; for (int r = 0; r < 8; ++r) acc[r] = bv;
#pragma unroll
        for (int c = 0; c < 5; ++c)
            acc = wmma32(frag_a(sX, 160, c * 32), frag_b_sw(sW1, 4, c, wid), acc);
        for (int r = 0; r < 8; ++r)
            sY[(r + 8 * hi) * 64 + n0 + n] = (_Float16)fmaxf(acc[r], 0.f);
    }
    __syncthreads();
    {
        float bv = bl2[n0 + n];
        v8f acc; for (int r = 0; r < 8; ++r) acc[r] = bv;
#pragma unroll
        for (int c = 0; c < 2; ++c)
            acc = wmma32(frag_a(sY, 64, c * 32), frag_b_sw(sW2, 4, c, wid), acc);
        // rows m = 8*hi + r; (fwd,bwd) of pair pp are rows (2q, 2q+1) of this lane
        for (int q = 0; q < 4; ++q) {
            int p = p0 + 4 * hi + q;
            if (p < NC2)
                outp[(b * NC2 + p) * 64 + n0 + n] = acc[2 * q] + acc[2 * q + 1];
        }
    }
}

// eplan = sinkhorn(log(max(straight, cross) + 1e-6)); whole 190x190 in LDS.
extern __shared__ float dynsm[];
__global__ void k_esi_eplan(const float* __restrict__ P, float* __restrict__ eplan) {
    __shared__ float Ps[400];
    __shared__ int ss[NC2], dd2[NC2];
    float* la = dynsm;                  // NC2*NC2 floats (144,400 B)
    int b = blockIdx.x, tid = threadIdx.x;   // 256
    for (int idx = tid; idx < 400; idx += 256) Ps[idx] = P[b * 400 + idx];
    if (tid < NC2) { int i, j; triu_ij(tid, i, j); ss[tid] = i; dd2[tid] = j; }
    __syncthreads();
    for (int idx = tid; idx < NC2 * NC2; idx += 256) {
        int p = idx / NC2, q = idx % NC2;
        float st = Ps[ss[p] * NNODE + ss[q]] * Ps[dd2[p] * NNODE + dd2[q]];
        float cr = Ps[ss[p] * NNODE + dd2[q]] * Ps[dd2[p] * NNODE + ss[q]];
        la[idx] = __logf(fmaxf(st, cr) + 1e-6f) / TEMP_;
    }
    __syncthreads();
    for (int it = 0; it < ITERS_; ++it) {
        if (tid < NC2) {
            const int base = tid * NC2;
            float m = -INFINITY;
            for (int q = 0; q < NC2; ++q) m = fmaxf(m, la[base + q]);
            float s = 0.f;
            for (int q = 0; q < NC2; ++q) s += __expf(la[base + q] - m);
            float lse = m + __logf(s);
            for (int q = 0; q < NC2; ++q) la[base + q] -= lse;
        }
        __syncthreads();
        if (tid < NC2) {
            float m = -INFINITY;
            for (int p = 0; p < NC2; ++p) m = fmaxf(m, la[p * NC2 + tid]);
            float s = 0.f;
            for (int p = 0; p < NC2; ++p) s += __expf(la[p * NC2 + tid] - m);
            float lse = m + __logf(s);
            for (int p = 0; p < NC2; ++p) la[p * NC2 + tid] -= lse;
        }
        __syncthreads();
    }
    for (int idx = tid; idx < NC2 * NC2; idx += 256)
        eplan[b * NC2 * NC2 + idx] = __expf(la[idx]);
}

// dist[b] = sum_{p,q} eplan[p,q] * L1(qe[p], ce[q]); ce[b] staged in LDS.
__global__ void k_dist(const float* __restrict__ qe, const float* __restrict__ ce,
                       const float* __restrict__ eplan, float* __restrict__ out) {
    __shared__ float sc[NC2 * 64];
    __shared__ float qrow[64];
    __shared__ float part[256];
    int b = blockIdx.x, tid = threadIdx.x;   // 256
    for (int idx = tid; idx < NC2 * 64; idx += 256) sc[idx] = ce[b * NC2 * 64 + idx];
    float acc = 0.f;
    for (int p = 0; p < NC2; ++p) {
        __syncthreads();
        if (tid < 64) qrow[tid] = qe[(b * NC2 + p) * 64 + tid];
        __syncthreads();
        for (int q = tid; q < NC2; q += 256) {
            float l1 = 0.f;
#pragma unroll 8
            for (int k = 0; k < 64; ++k) l1 += fabsf(qrow[k] - sc[q * 64 + k]);
            acc += eplan[b * NC2 * NC2 + p * NC2 + q] * l1;
        }
    }
    part[tid] = acc;
    __syncthreads();
    for (int s = 128; s > 0; s >>= 1) {
        if (tid < s) part[tid] += part[tid + s];
        __syncthreads();
    }
    if (tid == 0) out[b] = part[0];
}

// ---------------- host orchestration ----------------

extern "C" void kernel_launch(void* const* d_in, const int* in_sizes, int n_in,
                              void* d_out, int out_size, void* d_ws, size_t ws_size,
                              hipStream_t stream) {
    (void)in_sizes; (void)n_in; (void)out_size; (void)ws_size;
    const float* nf   = (const float*)d_in[0];
    const float* ef   = (const float*)d_in[1];
    const float* qadj = (const float*)d_in[2];
    const float* cadj = (const float*)d_in[3];
    const int*   from = (const int*)d_in[4];
    const int*   to   = (const int*)d_in[5];
    const float* Wn  = (const float*)d_in[6];  const float* bn  = (const float*)d_in[7];
    const float* We  = (const float*)d_in[8];  const float* be  = (const float*)d_in[9];
    const float* Wm1 = (const float*)d_in[10]; const float* bm1 = (const float*)d_in[11];
    const float* Wm2 = (const float*)d_in[12]; const float* bm2 = (const float*)d_in[13];
    const float* Wu1 = (const float*)d_in[14]; const float* bu1 = (const float*)d_in[15];
    const float* Wu2 = (const float*)d_in[16]; const float* bu2 = (const float*)d_in[17];
    const float* Ws1 = (const float*)d_in[18]; const float* bs1 = (const float*)d_in[19];
    const float* Ws2 = (const float*)d_in[20]; const float* bs2 = (const float*)d_in[21];
    const float* Wl1 = (const float*)d_in[22]; const float* bl1 = (const float*)d_in[23];
    const float* Wl2 = (const float*)d_in[24]; const float* bl2 = (const float*)d_in[25];

    float* ws = (float*)d_ws;
    float* hA  = ws; ws += 1280 * 64;
    float* hB  = ws; ws += 1280 * 64;
    float* eE  = ws; ws += 2560 * 64;
    float* agg = ws; ws += 1280 * 64;
    float* Tq  = ws; ws += 32 * 400;
    float* Tc  = ws; ws += 32 * 400;
    float* P   = ws; ws += 32 * 400;
    float* qe  = ws; ws += 32 * NC2 * 64;
    float* ce  = ws; ws += 32 * NC2 * 64;
    float* epl = ws; ws += 32 * NC2 * NC2;

    k_embed<<<80, 128, 0, stream>>>(nf, Wn, bn, 16, hA);
    k_embed<<<160, 128, 0, stream>>>(ef, We, be, 8, eE);

    float* hc = hA; float* hn = hB;
    for (int it = 0; it < 5; ++it) {
        k_zero<<<(1280 * 64 + 255) / 256, 256, 0, stream>>>(agg, 1280 * 64);
        k_message<<<160, 128, 0, stream>>>(hc, eE, from, to, Wm1, bm1, Wm2, bm2, agg);
        k_update<<<80, 128, 0, stream>>>(hc, agg, Wu1, bu1, Wu2, bu2, hn);
        float* t = hc; hc = hn; hn = t;
    }

    k_scores<<<80, 64, 0, stream>>>(hc, Ws1, bs1, Ws2, bs2, Tq, Tc);
    k_cost_sinkhorn<<<32, 128, 0, stream>>>(Tq, Tc, P);
    k_pair_emb<<<dim3(24, 32, 2), 128, 0, stream>>>(hc, qadj, cadj, Wl1, bl1, Wl2, bl2, qe, ce);
    k_esi_eplan<<<32, 256, NC2 * NC2 * sizeof(float), stream>>>(P, epl);
    k_dist<<<32, 256, 0, stream>>>(qe, ce, epl, (float*)d_out);
}
